// Conv2d_NN_prev_86165633892640
// MI455X (gfx1250) — matrix operand order repaired and checked by reading, not compiled, not run on hardware
//
#include <hip/hip_runtime.h>
#include <math.h>

typedef __attribute__((ext_vector_type(16))) _Float16 v16h;
typedef __attribute__((ext_vector_type(8)))  _Float16 v8h;
typedef __attribute__((ext_vector_type(8)))  float    v8f;

#define BB   8
#define CC   64
#define HH   56
#define WW   56
#define NN   (HH*WW)     /* 3136 */
#define CIN  66
#define CPAD 96          /* pad 66 -> 96 = 3 k-steps of 32 for f16 WMMA */
#define KNN  9
#define EPSF 1e-12f
#define NTILES (NN/16)   /* 196 */
#define TILES_PER_STAGE 4
#define NSTAGES (NTILES/TILES_PER_STAGE)        /* 49 */
#define STAGE_BYTES (TILES_PER_STAGE*16*CPAD*2) /* 12288 */

// ---------------------------------------------------------------------------
// CDNA5 async global->LDS helpers (ASYNCcnt-tracked; ISA 08_async_tensor §4)
// ---------------------------------------------------------------------------
__device__ __forceinline__ void async_ld_b128(void* lds_ptr, const void* gptr) {
  uint32_t loff = (uint32_t)(uintptr_t)lds_ptr;  // flat[31:0] == LDS offset (ISA §10.2)
  asm volatile("global_load_async_to_lds_b128 %0, %1, off"
               :: "v"(loff), "v"(gptr) : "memory");
}
__device__ __forceinline__ void wait_async0() {
  asm volatile("s_wait_asynccnt 0x0" ::: "memory");
}

// ---------------------------------------------------------------------------
// Kernel 1: build normalized f16 token matrix Tn[B][N][CPAD]
// ---------------------------------------------------------------------------
__global__ void k_prep(const float* __restrict__ x, _Float16* __restrict__ Tn) {
  int t = blockIdx.x * blockDim.x + threadIdx.x;
  if (t >= BB * NN) return;
  int b = t / NN, n = t % NN;
  int h = n / WW, w = n % WW;
  float cr   = sqrtf((float)(h * h + w * w));
  float cinv = 1.0f / fmaxf(cr, EPSF);
  float c64 = (float)h * cinv, c65 = (float)w * cinv;

  const float* xp = x + (size_t)b * CC * NN + n;
  float v[CIN];
  float ss = 0.f;
#pragma unroll
  for (int c = 0; c < CC; ++c) { float f = xp[(size_t)c * NN]; v[c] = f; ss += f * f; }
  v[64] = c64; v[65] = c65;
  ss += c64 * c64 + c65 * c65;
  float inv = 1.0f / fmaxf(sqrtf(ss), EPSF);

  _Float16* o = Tn + (size_t)(b * NN + n) * CPAD;
#pragma unroll
  for (int c = 0; c < CIN; ++c) o[c] = (_Float16)(v[c] * inv);
#pragma unroll
  for (int c = CIN; c < CPAD; ++c) o[c] = (_Float16)0.0f;
}

// ---------------------------------------------------------------------------
// Kernel 2: effective weights  weff[c][k][o] = sum_i w2[o][i]*w1[i][c][k]
// ---------------------------------------------------------------------------
__global__ void k_weff(const float* __restrict__ w1, const float* __restrict__ b1,
                       const float* __restrict__ w2, const float* __restrict__ b2,
                       float* __restrict__ weff, float* __restrict__ ebias) {
  int t = blockIdx.x * blockDim.x + threadIdx.x;
  const int NW = CIN * KNN * CC;
  if (t < NW) {
    int o  = t % CC;
    int ck = t / CC;
    int k  = ck % KNN;
    int c  = ck / KNN;
    float s = 0.f;
    for (int i = 0; i < CIN; ++i)
      s += w2[o * CIN + i] * w1[(size_t)(i * CIN + c) * KNN + k];
    weff[t] = s;
  } else if (t < NW + CC) {
    int o = t - NW;
    float s = b2[o];
    for (int i = 0; i < CIN; ++i) s += w2[o * CIN + i] * b1[i];
    ebias[o] = s;
  }
}

// ---------------------------------------------------------------------------
// Kernel 3: fused cosine-similarity (WMMA f16->f32) + streaming top-9.
// 4 waves/block; wave owns a 16-row tile. Column tiles staged into LDS once
// per block via global_load_async_to_lds_b128 (double-buffered, 4 tiles/stage).
// Top-k: lane r scans cols 0..7 of each tile, lane r+16 scans cols 8..15;
// the two sorted-9 lists are merged at the end (lower index wins ties).
// ---------------------------------------------------------------------------
__global__ void __launch_bounds__(128) k_simtopk(const _Float16* __restrict__ Tn,
                                                 int* __restrict__ idxout) {
  __shared__ char  stageMem[2 * STAGE_BYTES];     // 24 KB double buffer
  __shared__ float sc[4][16][16];                 // per-wave score tile
  __shared__ float mv[4][16][2][KNN];             // merge staging (values)
  __shared__ int   mi[4][16][2][KNN];             // merge staging (indices)

  const int tid  = threadIdx.x;
  const int lane = tid & 31;
  const int wv   = tid >> 5;
  const int b    = blockIdx.x / (NTILES / 4);
  const int rowBase = ((blockIdx.x % (NTILES / 4)) * 4 + wv) * 16;
  const int half = lane >> 4;   // k-group / column-half
  const int sub  = lane & 15;   // row (A) / col (B)

  const _Float16* Tb = Tn + (size_t)b * NN * CPAD;

  // A fragments for this wave's 16 rows (held across the whole column loop)
  const _Float16* ar = Tb + (size_t)(rowBase + sub) * CPAD + half * 16;
  v8h l0 = *(const v8h*)(ar +  0), h0 = *(const v8h*)(ar +  8);
  v8h l1 = *(const v8h*)(ar + 32), h1 = *(const v8h*)(ar + 40);
  v8h l2 = *(const v8h*)(ar + 64), h2 = *(const v8h*)(ar + 72);
  v16h A0 = __builtin_shufflevector(l0, h0, 0,1,2,3,4,5,6,7,8,9,10,11,12,13,14,15);
  v16h A1 = __builtin_shufflevector(l1, h1, 0,1,2,3,4,5,6,7,8,9,10,11,12,13,14,15);
  v16h A2 = __builtin_shufflevector(l2, h2, 0,1,2,3,4,5,6,7,8,9,10,11,12,13,14,15);

  float tv[KNN];
  int   ti[KNN];
#pragma unroll
  for (int i = 0; i < KNN; ++i) { tv[i] = -3.0e38f; ti[i] = 0; }

  // prologue: issue async copy of stage 0 (identity memcpy, 6 x b128 / thread)
  {
    const char* g = (const char*)Tb;
#pragma unroll
    for (int j = 0; j < 6; ++j) {
      int off = (tid + j * 128) * 16;
      async_ld_b128(stageMem + off, g + off);
    }
  }

  for (int s = 0; s < NSTAGES; ++s) {
    wait_async0();
    __syncthreads();                       // stage s resident for all waves

    if (s + 1 < NSTAGES) {                 // issue next stage into other buffer
      const char* g = (const char*)Tb + (size_t)(s + 1) * STAGE_BYTES;
      char* nbuf = stageMem + ((s + 1) & 1) * STAGE_BYTES;
#pragma unroll
      for (int j = 0; j < 6; ++j) {
        int off = (tid + j * 128) * 16;
        async_ld_b128(nbuf + off, g + off);
      }
    }
    if (s + 2 < NSTAGES)                   // L2 prefetch two stages ahead
      __builtin_prefetch((const char*)Tb + (size_t)(s + 2) * STAGE_BYTES + tid * 96, 0, 1);

    const char* cbuf = stageMem + (s & 1) * STAGE_BYTES;
#pragma unroll
    for (int t = 0; t < TILES_PER_STAGE; ++t) {
      const int ct = s * TILES_PER_STAGE + t;
      const _Float16* bc = (const _Float16*)(cbuf + (t * 16 + sub) * (CPAD * 2) + half * 32);
      v16h B0 = *(const v16h*)(bc +  0);
      v16h B1 = *(const v16h*)(bc + 32);
      v16h B2 = *(const v16h*)(bc + 64);

      v8f acc = {};
      acc = __builtin_amdgcn_wmma_f32_16x16x32_f16(false, A0, false, B0, (short)0, acc, false, false);
      acc = __builtin_amdgcn_wmma_f32_16x16x32_f16(false, A1, false, B1, (short)0, acc, false, false);
      acc = __builtin_amdgcn_wmma_f32_16x16x32_f16(false, A2, false, B2, (short)0, acc, false, false);

      // wave-private scatter + scan: LDS is in-order per wave, no barrier
#pragma unroll
      for (int j = 0; j < 8; ++j) sc[wv][j + 8 * half][sub] = acc[j];

#pragma unroll
      for (int q = 0; q < 8; ++q) {
        int n2 = half * 8 + q;
        float v = sc[wv][sub][n2];
        if (v > tv[8]) {                   // strict > : earliest index on ties
          tv[8] = v; ti[8] = ct * 16 + n2;
#pragma unroll
          for (int st = 8; st >= 1; --st) {
            if (tv[st] > tv[st - 1]) {
              float a = tv[st]; tv[st] = tv[st - 1]; tv[st - 1] = a;
              int   q2 = ti[st]; ti[st] = ti[st - 1]; ti[st - 1] = q2;
            }
          }
        }
      }
    }
    __syncthreads();                       // all waves done reading stage s
  }

  // merge the two per-half lists (same wave: program order suffices)
#pragma unroll
  for (int j = 0; j < KNN; ++j) { mv[wv][sub][half][j] = tv[j]; mi[wv][sub][half][j] = ti[j]; }

  if (lane < 16) {
    const float* av = &mv[wv][lane][0][0];
    const int*   ai = &mi[wv][lane][0][0];
    const float* bv = &mv[wv][lane][1][0];
    const int*   bi = &mi[wv][lane][1][0];
    int* op = idxout + ((size_t)b * NN + rowBase + lane) * KNN;
    int ia = 0, ib = 0;
#pragma unroll
    for (int j = 0; j < KNN; ++j) {
      float va = av[ia], vb = bv[ib];
      int   xa = ai[ia], xb = bi[ib];
      bool takeA = (va > vb) || ((va == vb) && (xa < xb));
      op[j] = takeA ? xa : xb;
      if (takeA) ++ia; else ++ib;
    }
  }
}

// ---------------------------------------------------------------------------
// Kernel 4: gather 9 neighbors (raw features + recomputed coords), contract
// with weff, add ebias, write with pixel_shuffle(2).
// ---------------------------------------------------------------------------
__global__ void k_out(const float* __restrict__ x, const int* __restrict__ idx,
                      const float* __restrict__ weff, const float* __restrict__ ebias,
                      float* __restrict__ out) {
  int t = blockIdx.x * blockDim.x + threadIdx.x;
  if (t >= BB * NN) return;
  int b = t / NN, n = t % NN;
  int h = n / WW, w = n % WW;

  float acc[CC];
#pragma unroll
  for (int o = 0; o < CC; ++o) acc[o] = ebias[o];

  const int*   ip = idx + (size_t)t * KNN;
  const float* xb = x + (size_t)b * CC * NN;

  for (int k = 0; k < KNN; ++k) {
    int nb = ip[k];
    int nh = nb / WW, nw = nb % WW;
    float cr   = sqrtf((float)(nh * nh + nw * nw));
    float cinv = 1.0f / fmaxf(cr, EPSF);
    for (int c = 0; c < CC; ++c) {
      float f = xb[(size_t)c * NN + nb];
      const float* wp = weff + (size_t)(c * KNN + k) * CC;
#pragma unroll
      for (int o = 0; o < CC; ++o) acc[o] += f * wp[o];
    }
    {
      float f = (float)nh * cinv;
      const float* wp = weff + (size_t)(64 * KNN + k) * CC;
#pragma unroll
      for (int o = 0; o < CC; ++o) acc[o] += f * wp[o];
    }
    {
      float f = (float)nw * cinv;
      const float* wp = weff + (size_t)(65 * KNN + k) * CC;
#pragma unroll
      for (int o = 0; o < CC; ++o) acc[o] += f * wp[o];
    }
  }

  float* ob = out + (size_t)b * 16 * 112 * 112;
#pragma unroll
  for (int o = 0; o < CC; ++o) {
    int cc2 = o >> 2, r1 = (o >> 1) & 1, r2 = o & 1;
    ob[((size_t)cc2 * 112 + (2 * h + r1)) * 112 + (2 * w + r2)] = acc[o];
  }
}

// ---------------------------------------------------------------------------
extern "C" void kernel_launch(void* const* d_in, const int* in_sizes, int n_in,
                              void* d_out, int out_size, void* d_ws, size_t ws_size,
                              hipStream_t stream) {
  const float* x  = (const float*)d_in[0];
  const float* w1 = (const float*)d_in[1];
  const float* b1 = (const float*)d_in[2];
  const float* w2 = (const float*)d_in[3];
  const float* b2 = (const float*)d_in[4];
  float* out = (float*)d_out;

  char* ws = (char*)d_ws;
  size_t szTn  = (size_t)BB * NN * CPAD * sizeof(_Float16);      // ~4.8 MB
  size_t offIdx = (szTn + 255) & ~(size_t)255;
  size_t szIdx = (size_t)BB * NN * KNN * sizeof(int);            // ~0.9 MB
  size_t offW  = (offIdx + szIdx + 255) & ~(size_t)255;
  size_t szW   = (size_t)CIN * KNN * CC * sizeof(float);         // 152 KB
  size_t offE  = (offW + szW + 255) & ~(size_t)255;

  _Float16* Tn   = (_Float16*)(ws);
  int*      idxb = (int*)(ws + offIdx);
  float*    weff = (float*)(ws + offW);
  float*    ebia = (float*)(ws + offE);

  k_prep<<<(BB * NN + 255) / 256, 256, 0, stream>>>(x, Tn);
  k_weff<<<(CIN * KNN * CC + CC + 255) / 256, 256, 0, stream>>>(w1, b1, w2, b2, weff, ebia);
  k_simtopk<<<BB * (NTILES / 4), 128, 0, stream>>>(Tn, idxb);
  k_out<<<(BB * NN + 255) / 256, 256, 0, stream>>>(x, idxb, weff, ebia, out);
}
